// PAM_2920577761835
// MI455X (gfx1250) — matrix operand order repaired and checked by reading, not compiled, not run on hardware
//
#include <hip/hip_runtime.h>
#include <hip/hip_bf16.h>

typedef _Float16 v16h __attribute__((ext_vector_type(16)));
typedef _Float16 v8h  __attribute__((ext_vector_type(8)));
typedef _Float16 v2h  __attribute__((ext_vector_type(2)));
typedef float    v8f  __attribute__((ext_vector_type(8)));
typedef float    v4f  __attribute__((ext_vector_type(4)));

#define WMMA_F16(A, B, Cacc) \
  __builtin_amdgcn_wmma_f32_16x16x32_f16(false, (A), false, (B), (short)0, (Cacc), false, false)

#if __has_builtin(__builtin_amdgcn_rcpf)
#define FAST_RCP(x) __builtin_amdgcn_rcpf(x)
#else
#define FAST_RCP(x) (1.0f / (x))
#endif

static constexpr int Cc = 512;   // channels
static constexpr int Nn = 4096;  // H*W
static constexpr int CQ = 64;    // q/k channels
static constexpr int Bb = 4;     // batch

// ---------------------------------------------------------------------------
// Build a 16-bit WMMA A-fragment from an fp32 row-major weight row:
// lane<16 takes K {0..15}, lane>=16 takes K {16..31}.
// ---------------------------------------------------------------------------
__device__ inline v16h load_w_afrag(const float* wrow) {
  v4f w0 = *(const v4f*)(wrow);
  v4f w1 = *(const v4f*)(wrow + 4);
  v4f w2 = *(const v4f*)(wrow + 8);
  v4f w3 = *(const v4f*)(wrow + 12);
  v16h a;
#pragma unroll
  for (int i = 0; i < 4; ++i) {
    a[i]      = (_Float16)w0[i];
    a[4 + i]  = (_Float16)w1[i];
    a[8 + i]  = (_Float16)w2[i];
    a[12 + i] = (_Float16)w3[i];
  }
  return a;
}

// B-fragment from x (32 chans x 16 pixels), lane = pixel column.
__device__ inline v16h load_x_bfrag(const float* xp) {
  v16h bb;
#pragma unroll
  for (int e = 0; e < 16; ++e) bb[e] = (_Float16)xp[(size_t)e * Nn];
  return bb;
}

// ---------------------------------------------------------------------------
// Kernel 1: q/k projections via WMMA -> fp16 [n][o] layout.
// One wave computes 64(o) x 32(n); blockIdx.y selects q vs k.
// q is pre-scaled by log2(e) so the attention softmax can use native exp2.
// ---------------------------------------------------------------------------
__global__ void __launch_bounds__(256) proj_qk(
    const float* __restrict__ x,
    const float* __restrict__ Wq, const float* __restrict__ bq,
    const float* __restrict__ Wk, const float* __restrict__ bk,
    _Float16* __restrict__ qh, _Float16* __restrict__ kh) {
  int lane = threadIdx.x & 31;
  int wave = threadIdx.x >> 5;
  int id = blockIdx.x * 8 + wave;        // 512 wave-tiles
  int nt2 = id & 127;                    // 128 tiles of 32 n
  int b   = id >> 7;                     // 0..3
  const float* W    = blockIdx.y ? Wk : Wq;
  const float* bias = blockIdx.y ? bk : bq;
  _Float16*    dst  = blockIdx.y ? kh : qh;
  float scale = blockIdx.y ? 1.0f : 1.44269504f;
  int nbase = nt2 * 32;
  int l15 = lane & 15;
  int lh  = (lane < 16) ? 0 : 1;

  v8f acc[4][2];
#pragma unroll
  for (int oo = 0; oo < 4; ++oo)
#pragma unroll
    for (int nn = 0; nn < 2; ++nn) acc[oo][nn] = (v8f){};

  for (int c = 0; c < Cc; c += 32) {
    v16h bf0 = load_x_bfrag(x + ((size_t)(b * Cc + c + lh * 16)) * Nn + nbase + l15);
    v16h bf1 = load_x_bfrag(x + ((size_t)(b * Cc + c + lh * 16)) * Nn + nbase + 16 + l15);
#pragma unroll
    for (int oo = 0; oo < 4; ++oo) {
      v16h a = load_w_afrag(W + (oo * 16 + l15) * Cc + c + lh * 16);
      acc[oo][0] = WMMA_F16(a, bf0, acc[oo][0]);
      acc[oo][1] = WMMA_F16(a, bf1, acc[oo][1]);
    }
  }
#pragma unroll
  for (int oo = 0; oo < 4; ++oo) {
#pragma unroll
    for (int nn = 0; nn < 2; ++nn) {
#pragma unroll
      for (int r = 0; r < 8; ++r) {
        int o = oo * 16 + r + lh * 8;
        int n = nbase + nn * 16 + l15;
        dst[((size_t)(b * Nn + n)) * CQ + o] =
            (_Float16)((acc[oo][nn][r] + bias[o]) * scale);
      }
    }
  }
}

// ---------------------------------------------------------------------------
// Kernel 2: v projection via WMMA -> vt[b][o][n] fp16 (channel-major).
// One wave computes 64(o) x 32(n): two x B-fragments shared by 4 o-subtiles,
// four W A-fragments shared by 2 n-subtiles -> 48 VMEM per 8 WMMAs.
// ---------------------------------------------------------------------------
__global__ void __launch_bounds__(256) proj_v(const float* __restrict__ x,
                                              const float* __restrict__ Wv,
                                              const float* __restrict__ bv,
                                              _Float16* __restrict__ vt) {
  int lane = threadIdx.x & 31;
  int wave = threadIdx.x >> 5;
  int id = blockIdx.x * 8 + wave;        // 4096 wave-tiles
  int nt2 = id & 127;                    // 128 tiles of 32 n
  int og  = id >> 7;                     // 0..31 : 8 o-groups x 4 batches
  int b = og >> 3;
  int obase = (og & 7) * 64;
  int nbase = nt2 * 32;
  int l15 = lane & 15;
  int lh  = (lane < 16) ? 0 : 1;

  v8f acc[4][2];
#pragma unroll
  for (int oo = 0; oo < 4; ++oo)
#pragma unroll
    for (int nn = 0; nn < 2; ++nn) acc[oo][nn] = (v8f){};

  for (int c = 0; c < Cc; c += 32) {
    v16h bf0 = load_x_bfrag(x + ((size_t)(b * Cc + c + lh * 16)) * Nn + nbase + l15);
    v16h bf1 = load_x_bfrag(x + ((size_t)(b * Cc + c + lh * 16)) * Nn + nbase + 16 + l15);
#pragma unroll
    for (int oo = 0; oo < 4; ++oo) {
      v16h a = load_w_afrag(Wv + (obase + oo * 16 + l15) * Cc + c + lh * 16);
      acc[oo][0] = WMMA_F16(a, bf0, acc[oo][0]);
      acc[oo][1] = WMMA_F16(a, bf1, acc[oo][1]);
    }
  }
#pragma unroll
  for (int oo = 0; oo < 4; ++oo) {
#pragma unroll
    for (int nn = 0; nn < 2; ++nn) {
#pragma unroll
      for (int r = 0; r < 8; ++r) {
        int orow = obase + oo * 16 + r + lh * 8;
        float val = acc[oo][nn][r] + bv[orow];
        vt[((size_t)(b * Cc + orow)) * Nn + nbase + nn * 16 + l15] =
            (_Float16)val;
      }
    }
  }
}

// ---------------------------------------------------------------------------
// Kernel 3: flash attention, 64-key tiles, base-2 softmax.
// One block = one 16-query tile; 2 waves own 256-channel V/O slices
// (16 f32 accumulator fragments = 128 VGPRs). Row sums via WMMA against an
// all-ones B fragment; P transposed D->A through wave-private LDS as packed
// f16 pairs; per-row corr gathered into a v8f for packed-f32 rescale.
// ---------------------------------------------------------------------------
__global__ void __launch_bounds__(64) flash_attn(
    const _Float16* __restrict__ qh, const _Float16* __restrict__ kh,
    const _Float16* __restrict__ vt, const float* __restrict__ x,
    const float* __restrict__ gamma, float* __restrict__ out) {
  // 2 waves x 16 rows x 32 dwords (64 packed f16 keys) = 4KB
  __shared__ __align__(32) unsigned int pbuf32[2 * 16 * 32];
  int lane = threadIdx.x & 31;
  int wave = threadIdx.x >> 5;
  int b  = blockIdx.x >> 8;
  int m0 = (blockIdx.x & 255) * 16;     // query-row base
  int chbase = wave * 256;              // this wave's V/O channel slice
  int l15 = lane & 15;
  int lh  = (lane < 16) ? 0 : 1;
  v2h* pw = (v2h*)(pbuf32 + wave * 512);

  // Q A-fragments (rows m0..m0+15, chans 0..31 / 32..63)
  const _Float16* qrow = qh + ((size_t)(b * Nn + m0 + l15)) * CQ;
  v8h q0lo = *(const v8h*)(qrow + lh * 16);
  v8h q0hi = *(const v8h*)(qrow + 8 + lh * 16);
  v8h q1lo = *(const v8h*)(qrow + 32 + lh * 16);
  v8h q1hi = *(const v8h*)(qrow + 40 + lh * 16);
  v16h qa0 = __builtin_shufflevector(q0lo, q0hi, 0,1,2,3,4,5,6,7,8,9,10,11,12,13,14,15);
  v16h qa1 = __builtin_shufflevector(q1lo, q1hi, 0,1,2,3,4,5,6,7,8,9,10,11,12,13,14,15);

  v16h ones;
#pragma unroll
  for (int i = 0; i < 16; ++i) ones[i] = (_Float16)1.0f;

  v8f acc[16];
#pragma unroll
  for (int i = 0; i < 16; ++i) acc[i] = (v8f){};
  v8f accl = {};                         // row sums l via WMMA
  float mrow[8];
#pragma unroll
  for (int r = 0; r < 8; ++r) mrow[r] = -1e30f;

  // loop-carried base pointers; all tile offsets become 24-bit immediates
  const _Float16* kptr = kh + ((size_t)(b * Nn + l15)) * CQ + lh * 16;
  const _Float16* vptr = vt + ((size_t)(b * Cc + chbase + l15)) * Nn + lh * 8;

#pragma unroll 1
  for (int kt = 0; kt < Nn; kt += 64) {
    // ---- S = Q * K^T for 64 keys (four 16-col fragments) ----
    v8f s[4];
#pragma unroll
    for (int j = 0; j < 4; ++j) {
      v16h kb0 = *(const v16h*)(kptr + j * 16 * CQ);       // chans 0..31
      v16h kb1 = *(const v16h*)(kptr + j * 16 * CQ + 32);  // chans 32..63
      v8f sj = {};
      sj = WMMA_F16(qa0, kb0, sj);
      sj = WMMA_F16(qa1, kb1, sj);
      s[j] = sj;
    }
    kptr += 64 * CQ;

    // ---- online softmax (base-2): max reduce, emit packed P to LDS ----
    v8f corrv;
#pragma unroll
    for (int r = 0; r < 8; ++r) {
      float t0 = s[0][r], t1 = s[1][r], t2 = s[2][r], t3 = s[3][r];
      float mx = fmaxf(fmaxf(t0, t1), fmaxf(t2, t3));
      mx = fmaxf(mx, __shfl_xor(mx, 1, 32));
      mx = fmaxf(mx, __shfl_xor(mx, 2, 32));
      mx = fmaxf(mx, __shfl_xor(mx, 4, 32));
      mx = fmaxf(mx, __shfl_xor(mx, 8, 32));
      float mnew = fmaxf(mrow[r], mx);
      corrv[r] = exp2f(mrow[r] - mnew);   // native v_exp_f32
      mrow[r] = mnew;
      float p0 = exp2f(t0 - mnew);
      float p1 = exp2f(t1 - mnew);
      float p2 = exp2f(t2 - mnew);
      float p3 = exp2f(t3 - mnew);
      int prow = r + lh * 8;               // D-layout row for this lane
      v2h pk01; pk01[0] = (_Float16)p0; pk01[1] = (_Float16)p1;
      v2h pk23; pk23[0] = (_Float16)p2; pk23[1] = (_Float16)p3;
      pw[(prow * 2 + 0) * 16 + l15] = pk01;   // keys kt+0..31 packed
      pw[(prow * 2 + 1) * 16 + l15] = pk23;   // keys kt+32..63 packed
    }

    // ---- packed-f32 rescale of all accumulators ----
    accl *= corrv;
#pragma unroll
    for (int cb = 0; cb < 16; ++cb) acc[cb] *= corrv;

    // ---- P: D-layout -> two A-fragments (deinterleave packed pairs) ----
    v16h raw0 = *(const v16h*)(pw + (l15 * 2 + 0) * 16 + lh * 8);
    v16h raw1 = *(const v16h*)(pw + (l15 * 2 + 1) * 16 + lh * 8);
    v16h pa0 = __builtin_shufflevector(raw0, raw0,
                 0,2,4,6,8,10,12,14, 1,3,5,7,9,11,13,15);
    v16h pa1 = __builtin_shufflevector(raw1, raw1,
                 0,2,4,6,8,10,12,14, 1,3,5,7,9,11,13,15);

    // ---- row sums l += P * 1 (WMMA, consistent with f16 P numerator) ----
    accl = WMMA_F16(pa0, ones, accl);
    accl = WMMA_F16(pa1, ones, accl);

    // ---- O += P * V over this wave's 16 channel blocks ----
#pragma unroll
    for (int cb = 0; cb < 16; ++cb) {
      v8h v0a = *(const v8h*)(vptr + cb * 16 * Nn);
      v8h v0b = *(const v8h*)(vptr + cb * 16 * Nn + 16);
      v8h v1a = *(const v8h*)(vptr + cb * 16 * Nn + 32);
      v8h v1b = *(const v8h*)(vptr + cb * 16 * Nn + 48);
      v16h vb0 = __builtin_shufflevector(v0a, v0b, 0,1,2,3,4,5,6,7,8,9,10,11,12,13,14,15);   // keys kt+0..31
      v16h vb1 = __builtin_shufflevector(v1a, v1b, 0,1,2,3,4,5,6,7,8,9,10,11,12,13,14,15);   // keys kt+32..63
      acc[cb] = WMMA_F16(pa0, vb0, acc[cb]);
      acc[cb] = WMMA_F16(pa1, vb1, acc[cb]);
    }
    vptr += 64;
  }

  // ---- epilogue: out = gamma * ctx + x  (native rcp, no divides) ----
  float g = gamma[0];
  float sc[8];
#pragma unroll
  for (int r = 0; r < 8; ++r) sc[r] = g * FAST_RCP(accl[r]);
#pragma unroll
  for (int cb = 0; cb < 16; ++cb) {
#pragma unroll
    for (int r = 0; r < 8; ++r) {
      int row = m0 + r + lh * 8;
      int ch  = chbase + cb * 16 + l15;
      size_t idx = ((size_t)(b * Cc + ch)) * Nn + row;
      out[idx] = fmaf(acc[cb][r], sc[r], x[idx]);
    }
  }
}

// ---------------------------------------------------------------------------
extern "C" void kernel_launch(void* const* d_in, const int* in_sizes, int n_in,
                              void* d_out, int out_size, void* d_ws, size_t ws_size,
                              hipStream_t stream) {
  const float* x     = (const float*)d_in[0];
  const float* Wq    = (const float*)d_in[1];
  const float* bq    = (const float*)d_in[2];
  const float* Wk    = (const float*)d_in[3];
  const float* bk    = (const float*)d_in[4];
  const float* Wv    = (const float*)d_in[5];
  const float* bv    = (const float*)d_in[6];
  const float* gamma = (const float*)d_in[7];

  char* ws = (char*)d_ws;
  _Float16* qh = (_Float16*)(ws);                           // 2 MB
  _Float16* kh = (_Float16*)(ws + (size_t)2 * 1024 * 1024); // 2 MB
  _Float16* vt = (_Float16*)(ws + (size_t)4 * 1024 * 1024); // 16 MB

  proj_qk<<<dim3(64, 2), 256, 0, stream>>>(x, Wq, bq, Wk, bk, qh, kh);
  proj_v<<<512, 256, 0, stream>>>(x, Wv, bv, vt);
  flash_attn<<<Bb * (Nn / 16), 64, 0, stream>>>(qh, kh, vt, x, gamma,
                                                (float*)d_out);
}
